// SAGPool_64235530879311
// MI455X (gfx1250) — matrix operand order repaired and checked by reading, not compiled, not run on hardware
//
#include <hip/hip_runtime.h>
#include <hip/hip_bf16.h>
#include <math.h>

// ---------------------------------------------------------------------------
// SAGPool GNN forward for MI455X (gfx1250, wave32, WMMA).
// Dense matmuls use v_wmma_f32_16x16x32_f16 (f32 accum, f16 inputs).
// Irregular graph ops (scatter, top-k, gather, remap) are plain VALU/VMEM.
// ---------------------------------------------------------------------------

typedef __attribute__((ext_vector_type(16))) _Float16 v16h;
typedef __attribute__((ext_vector_type(8)))  float    v8f;

#define HD 128          // hidden / feature dim
#define NLAY 5          // JK concat layers

union Frag16 { v16h v; _Float16 h[16]; };

__device__ __forceinline__ void load8f(const float* __restrict__ p, float* d) {
  const float4 u0 = *(const float4*)(p);
  const float4 u1 = *(const float4*)(p + 4);
  d[0]=u0.x; d[1]=u0.y; d[2]=u0.z; d[3]=u0.w;
  d[4]=u1.x; d[5]=u1.y; d[6]=u1.z; d[7]=u1.w;
}

// Accumulate acc += (scale*Arow[0:K]) @ Wcol, K = 32*ksteps.
// A-fragment layout (16-bit A 16x32): lane holds row M=lane%16,
// element e -> K = e + (lane/16)*16 + kbase.
// B-fragment (32x16): lane holds col N=lane%16, element e -> K = e + (lane/16)*16.
// Wcol points at col-major f16 weights for this lane's output column.
__device__ __forceinline__ v8f wmma_acc(const float* __restrict__ arow, float scale,
                                        const _Float16* __restrict__ wcol,
                                        int half, int ksteps, v8f acc) {
  for (int ks = 0; ks < ksteps; ++ks) {
    const int kbase = ks * 32;
    float av[16];
    load8f(arow + kbase + half * 16,       av);
    load8f(arow + kbase + half * 16 + 8,   av + 8);
    Frag16 af;
#pragma unroll
    for (int e = 0; e < 16; ++e) af.h[e] = (_Float16)(av[e] * scale);
    const v16h bf = *(const v16h*)(wcol + kbase + half * 16);
    acc = __builtin_amdgcn_wmma_f32_16x16x32_f16(false, af.v, false, bf,
                                                 (short)0, acc, false, false);
  }
  return acc;
}

// out[M,128] = relu( (agg/max(deg,1)) @ Wr + X @ Ws + b )
// one block = 8 waves; block -> 16 rows, wave w -> columns [16w,16w+16)
__global__ void __launch_bounds__(256) conv_gemm_k(
    const float* __restrict__ agg, const float* __restrict__ deg,
    const float* __restrict__ X,
    const _Float16* __restrict__ WrT, const _Float16* __restrict__ WsT,
    const float* __restrict__ bias, float* __restrict__ out) {
  const int wave = threadIdx.x >> 5;
  const int lane = threadIdx.x & 31;
  const int lrow = lane & 15;
  const int half = lane >> 4;
  const int row  = blockIdx.x * 16 + lrow;
  const int col  = wave * 16 + lrow;
  const float invd = 1.0f / fmaxf(deg[row], 1.0f);
  v8f acc = {};
  acc = wmma_acc(agg + (size_t)row * HD, invd, WrT + (size_t)col * HD, half, 4, acc);
  acc = wmma_acc(X   + (size_t)row * HD, 1.0f, WsT + (size_t)col * HD, half, 4, acc);
  const float b = bias[col];
#pragma unroll
  for (int v = 0; v < 8; ++v) {
    const int r = blockIdx.x * 16 + v + half * 8;   // C-frag: M = v + (lane/16)*8
    out[(size_t)r * HD + col] = fmaxf(acc[v] + b, 0.0f);
  }
}

// out[M,128] = relu( A[M,K] @ W + b ), W pre-transposed f16 [128][K]
__global__ void __launch_bounds__(256) gemm_relu_k(
    const float* __restrict__ A, const _Float16* __restrict__ WT,
    const float* __restrict__ bias, float* __restrict__ out, int K, int lda) {
  const int wave = threadIdx.x >> 5;
  const int lane = threadIdx.x & 31;
  const int lrow = lane & 15;
  const int half = lane >> 4;
  const int row  = blockIdx.x * 16 + lrow;
  const int col  = wave * 16 + lrow;
  v8f acc = {};
  acc = wmma_acc(A + (size_t)row * lda, 1.0f, WT + (size_t)col * K, half, K / 32, acc);
  const float b = bias[col];
#pragma unroll
  for (int v = 0; v < 8; ++v) {
    const int r = blockIdx.x * 16 + v + half * 8;
    out[(size_t)r * HD + col] = fmaxf(acc[v] + b, 0.0f);
  }
}

// weight convert+transpose: W[K,N] f32 -> WT[N,K] f16
__global__ void wcvt_k(const float* __restrict__ W, _Float16* __restrict__ WT,
                       int K, int N) {
  const int i = blockIdx.x * 256 + threadIdx.x;
  if (i >= K * N) return;
  const int k = i / N, n = i % N;
  WT[(size_t)n * K + k] = (_Float16)W[i];
}

__global__ void fill_k(float* __restrict__ p, float v, size_t n) {
  size_t i = (size_t)blockIdx.x * blockDim.x + threadIdx.x;
  const size_t step = (size_t)gridDim.x * blockDim.x;
  for (; i < n; i += step) p[i] = v;
}

__global__ void edge_init_k(const int* __restrict__ ei, int* __restrict__ esrc,
                            int* __restrict__ edst, float* __restrict__ emask, int E) {
  const int i = blockIdx.x * 256 + threadIdx.x;
  if (i >= E) return;
  esrc[i] = ei[i];
  edst[i] = ei[E + i];
  emask[i] = 1.0f;
}

// agg[dst] += x[src]*mask : block per edge, thread per channel
__global__ void __launch_bounds__(128) scatter_feat_k(
    const float* __restrict__ X, const int* __restrict__ src,
    const int* __restrict__ dst, const float* __restrict__ mask,
    float* __restrict__ agg) {
  const int e = blockIdx.x;
  const float m = mask[e];
  if (m == 0.0f) return;
  const int s = src[e], d = dst[e];
  const int c = threadIdx.x;
  atomicAdd(&agg[(size_t)d * HD + c], X[(size_t)s * HD + c] * m);
}

__global__ void scatter_deg_k(const int* __restrict__ dst,
                              const float* __restrict__ mask,
                              float* __restrict__ deg, int E) {
  const int i = blockIdx.x * 256 + threadIdx.x;
  if (i < E) atomicAdd(&deg[dst[i]], mask[i]);
}

// score = (agg/max(deg,1)) . wr + x . ws + b   (GraphConv H->1, no relu)
__global__ void score_k(const float* __restrict__ agg, const float* __restrict__ deg,
                        const float* __restrict__ X, const float* __restrict__ wr,
                        const float* __restrict__ ws, const float* __restrict__ b,
                        float* __restrict__ score, int n) {
  const int i = blockIdx.x * 256 + threadIdx.x;
  if (i >= n) return;
  const float invd = 1.0f / fmaxf(deg[i], 1.0f);
  const float* a = agg + (size_t)i * HD;
  const float* x = X + (size_t)i * HD;
  float s = b[0];
  for (int c = 0; c < HD; ++c) s += a[c] * invd * wr[c] + x[c] * ws[c];
  score[i] = s;
}

// per-graph top-k via rank (rank == descending-order position, ties by index,
// matching jax.lax.top_k). rank<k nodes keep new local index == rank.
__global__ void __launch_bounds__(128) topk_k(const float* __restrict__ score,
                                              int* __restrict__ node_map,
                                              int* __restrict__ inv_perm,
                                              int npg, int k) {
  __shared__ float s[512];
  const int g = blockIdx.x;
  const float* sc = score + (size_t)g * npg;
  for (int j = threadIdx.x; j < npg; j += blockDim.x) s[j] = sc[j];
  __syncthreads();
  for (int j = threadIdx.x; j < npg; j += blockDim.x) {
    const float mys = s[j];
    int rank = 0;
    for (int t = 0; t < npg; ++t) {
      const float o = s[t];
      rank += (o > mys) || (o == mys && t < j);
    }
    if (rank < k) {
      node_map[(size_t)g * npg + j] = g * k + rank;
      inv_perm[(size_t)g * k + rank] = g * npg + j;
    } else {
      node_map[(size_t)g * npg + j] = -1;
    }
  }
}

// new_x[r] = x[perm[r]] * tanh(score[perm[r]])
__global__ void __launch_bounds__(128) gather_k(const float* __restrict__ X,
                                                const float* __restrict__ score,
                                                const int* __restrict__ inv_perm,
                                                float* __restrict__ out) {
  const int r = blockIdx.x;
  const int old = inv_perm[r];
  const float t = tanhf(score[old]);
  out[(size_t)r * HD + threadIdx.x] = X[(size_t)old * HD + threadIdx.x] * t;
}

__global__ void remap_k(int* __restrict__ src, int* __restrict__ dst,
                        float* __restrict__ mask, const int* __restrict__ nmap, int E) {
  const int i = blockIdx.x * 256 + threadIdx.x;
  if (i >= E) return;
  const int ns = nmap[src[i]];
  const int nd = nmap[dst[i]];
  float m = mask[i];
  if (ns < 0 || nd < 0) m = 0.0f;
  mask[i] = m;
  src[i] = ns < 0 ? 0 : ns;
  dst[i] = nd < 0 ? 0 : nd;
}

// per-graph mean over npg nodes -> xs_all[g, layer*H + c]
__global__ void __launch_bounds__(128) meanpool_k(const float* __restrict__ X,
                                                  float* __restrict__ xs_all,
                                                  int npg, int col_off, int ldx) {
  const int g = blockIdx.x;
  const int c = threadIdx.x;
  const float* p = X + (size_t)g * npg * HD + c;
  float s = 0.0f;
  for (int j = 0; j < npg; ++j) s += p[(size_t)j * HD];
  xs_all[(size_t)g * ldx + col_off + c] = s / (float)npg;
}

// lin2 [128,2] + log_softmax
__global__ void head_k(const float* __restrict__ hb, const float* __restrict__ w2,
                       const float* __restrict__ b2, float* __restrict__ out, int B) {
  const int g = blockIdx.x * blockDim.x + threadIdx.x;
  if (g >= B) return;
  float l0 = b2[0], l1 = b2[1];
  const float* h = hb + (size_t)g * HD;
  for (int c = 0; c < HD; ++c) {
    l0 += h[c] * w2[c * 2 + 0];
    l1 += h[c] * w2[c * 2 + 1];
  }
  const float m = fmaxf(l0, l1);
  const float lse = m + logf(expf(l0 - m) + expf(l1 - m));
  out[g * 2 + 0] = l0 - lse;
  out[g * 2 + 1] = l1 - lse;
}

extern "C" void kernel_launch(void* const* d_in, const int* in_sizes, int n_in,
                              void* d_out, int out_size, void* d_ws, size_t ws_size,
                              hipStream_t stream) {
  (void)n_in; (void)ws_size;
  const float* x_in  = (const float*)d_in[0];
  const int*   ei    = (const int*)d_in[1];
  const float* c1_wr = (const float*)d_in[3];
  const float* c1_ws = (const float*)d_in[4];
  const float* c1_b  = (const float*)d_in[5];
  const float* cs_wr = (const float*)d_in[6];
  const float* cs_ws = (const float*)d_in[7];
  const float* cs_b  = (const float*)d_in[8];
  const float* p_wr  = (const float*)d_in[9];
  const float* p_ws  = (const float*)d_in[10];
  const float* p_b   = (const float*)d_in[11];
  const float* l1_w  = (const float*)d_in[12];
  const float* l1_b  = (const float*)d_in[13];
  const float* l2_w  = (const float*)d_in[14];
  const float* l2_b  = (const float*)d_in[15];

  const int N0 = in_sizes[0] / HD;     // 102400
  const int E  = in_sizes[1] / 2;      // 1638400
  const int B  = out_size / 2;         // 256 graphs
  const int HH = HD * HD;
  const int K1 = NLAY * HD;            // 640 (JK concat width)

  // ---- workspace carve (~180 MB) ----
  char* w = (char*)d_ws;
  auto alloc = [&](size_t bytes) {
    char* p = w; w += (bytes + 255) & ~(size_t)255; return p;
  };
  float* bufA   = (float*)alloc((size_t)N0 * HD * 4);
  float* bufB   = (float*)alloc((size_t)N0 * HD * 4);
  float* agg    = (float*)alloc((size_t)N0 * HD * 4);
  float* deg    = (float*)alloc((size_t)N0 * 4);
  float* scor   = (float*)alloc((size_t)N0 * 4);
  int*   esrc   = (int*)alloc((size_t)E * 4);
  int*   edst   = (int*)alloc((size_t)E * 4);
  float* emask  = (float*)alloc((size_t)E * 4);
  int*   nmap   = (int*)alloc((size_t)N0 * 4);
  int*   iperm  = (int*)alloc((size_t)N0 * 4);
  float* xs_all = (float*)alloc((size_t)B * K1 * 4);
  float* hbuf   = (float*)alloc((size_t)B * HD * 4);
  _Float16* wt  = (_Float16*)alloc((size_t)(10 * HH + K1 * HD) * 2);
  _Float16* wtl1 = wt + 10 * HH;

  // ---- weight conversion (f32 -> f16, transposed to [N][K]) ----
  const int wblk = (HH + 255) / 256;
  wcvt_k<<<wblk, 256, 0, stream>>>(c1_wr, wt + 0 * HH, HD, HD);
  wcvt_k<<<wblk, 256, 0, stream>>>(c1_ws, wt + 1 * HH, HD, HD);
  for (int l = 0; l < 4; ++l) {
    wcvt_k<<<wblk, 256, 0, stream>>>(cs_wr + (size_t)l * HH, wt + (2 + 2 * l) * HH, HD, HD);
    wcvt_k<<<wblk, 256, 0, stream>>>(cs_ws + (size_t)l * HH, wt + (3 + 2 * l) * HH, HD, HD);
  }
  wcvt_k<<<(K1 * HD + 255) / 256, 256, 0, stream>>>(l1_w, wtl1, K1, HD);

  // ---- edges ----
  edge_init_k<<<(E + 255) / 256, 256, 0, stream>>>(ei, esrc, edst, emask, E);

  auto zero = [&](float* p, size_t n) {
    fill_k<<<1024, 256, 0, stream>>>(p, 0.0f, n);
  };
  auto run_conv = [&](const float* xin, float* xout, const _Float16* wr,
                      const _Float16* wsp, const float* bias, int n) {
    zero(agg, (size_t)n * HD);
    zero(deg, (size_t)n);
    scatter_feat_k<<<E, 128, 0, stream>>>(xin, esrc, edst, emask, agg);
    scatter_deg_k<<<(E + 255) / 256, 256, 0, stream>>>(edst, emask, deg, E);
    conv_gemm_k<<<n / 16, 256, 0, stream>>>(agg, deg, xin, wr, wsp, bias, xout);
  };
  auto run_meanpool = [&](const float* xin, int npg, int layer) {
    meanpool_k<<<B, HD, 0, stream>>>(xin, xs_all, npg, layer * HD, K1);
  };
  auto run_pool = [&](const float* xin, float* xout, int j, int n, int npg, int k) {
    zero(agg, (size_t)n * HD);                 // deg unchanged since the conv
    scatter_feat_k<<<E, 128, 0, stream>>>(xin, esrc, edst, emask, agg);
    score_k<<<(n + 255) / 256, 256, 0, stream>>>(agg, deg, xin, p_wr + j * HD,
                                                 p_ws + j * HD, p_b + j, scor, n);
    topk_k<<<B, 128, 0, stream>>>(scor, nmap, iperm, npg, k);
    gather_k<<<B * k, HD, 0, stream>>>(xin, scor, iperm, xout);
    remap_k<<<(E + 255) / 256, 256, 0, stream>>>(esrc, edst, emask, nmap, E);
  };

  int n = N0;
  int npg = N0 / B;                           // 400

  // conv1
  run_conv(x_in, bufA, wt + 0 * HH, wt + 1 * HH, c1_b, n);
  run_meanpool(bufA, npg, 0);
  // convs[0]
  run_conv(bufA, bufB, wt + 2 * HH, wt + 3 * HH, cs_b + 0 * HD, n);
  run_meanpool(bufB, npg, 1);
  // pool 0 (k = ceil(0.8*npg))
  int k0 = (npg * 4 + 4) / 5;                 // 320
  run_pool(bufB, bufA, 0, n, npg, k0);
  n = B * k0; npg = k0;
  // convs[1]
  run_conv(bufA, bufB, wt + 4 * HH, wt + 5 * HH, cs_b + 1 * HD, n);
  run_meanpool(bufB, npg, 2);
  // convs[2]
  run_conv(bufB, bufA, wt + 6 * HH, wt + 7 * HH, cs_b + 2 * HD, n);
  run_meanpool(bufA, npg, 3);
  // pool 1
  int k1 = (npg * 4 + 4) / 5;                 // 256
  run_pool(bufA, bufB, 1, n, npg, k1);
  n = B * k1; npg = k1;
  // convs[3]
  run_conv(bufB, bufA, wt + 8 * HH, wt + 9 * HH, cs_b + 3 * HD, n);
  run_meanpool(bufA, npg, 4);

  // JK concat -> lin1 (WMMA GEMM, K=640) -> head
  gemm_relu_k<<<B / 16, 256, 0, stream>>>(xs_all, wtl1, l1_b, hbuf, K1, K1);
  head_k<<<1, B, 0, stream>>>(hbuf, l2_w, l2_b, (float*)d_out, B);
}
